// MoE_9500467658832
// MI455X (gfx1250) — matrix operand order repaired and checked by reading, not compiled, not run on hardware
//
#include <hip/hip_runtime.h>
#include <hip/hip_bf16.h>

#define NT 8192
#define DM 1024
#define DH 4096
#define NE 8
#define TK 2

typedef __attribute__((ext_vector_type(16))) __bf16 v16bf;
typedef __attribute__((ext_vector_type(8)))  __bf16 v8bf;
typedef __attribute__((ext_vector_type(8)))  float  v8f;

__device__ __forceinline__ unsigned short f2bf(float f) {
  unsigned u = __float_as_uint(f);
  u += 0x7FFFu + ((u >> 16) & 1u);   // round-to-nearest-even
  return (unsigned short)(u >> 16);
}

__device__ __forceinline__ unsigned ldsaddr(const void* p) {
  return (unsigned)(reinterpret_cast<uintptr_t>(p));   // low 32 bits = LDS byte addr
}

// Issue two b128 async global->LDS copies (32B per lane); INST_OFFSET moves
// both the LDS and global pointers per CDNA5 ISA 08 §4.4.
__device__ __forceinline__ void async_cp32(unsigned lds, const unsigned short* sbase,
                                           unsigned voff) {
  asm volatile("global_load_async_to_lds_b128 %0, %1, %2 offset:0"
               :: "v"(lds), "v"(voff), "s"(sbase) : "memory");
  asm volatile("global_load_async_to_lds_b128 %0, %1, %2 offset:16"
               :: "v"(lds), "v"(voff), "s"(sbase) : "memory");
}

// ---------------- init ----------------
__global__ void k_init(int* cnt, int* fill) {
  int t = threadIdx.x;
  if (t < NE) { cnt[t] = 0; fill[t] = 0; }
}

// ---------------- fp32 -> bf16 convert (x) ----------------
__global__ void k_convert(const float* __restrict__ src,
                          unsigned short* __restrict__ dst, int n) {
  int stride = gridDim.x * blockDim.x;
  for (int i = blockIdx.x * blockDim.x + threadIdx.x; i * 4 < n; i += stride) {
    float4 v = ((const float4*)src)[i];
    ushort4 o;
    o.x = f2bf(v.x); o.y = f2bf(v.y); o.z = f2bf(v.z); o.w = f2bf(v.w);
    ((ushort4*)dst)[i] = o;
  }
}

// ------- fp32 [E][R][C] -> bf16 transposed [E][C][R] (LDS-tiled 32x32) -------
__global__ __launch_bounds__(256) void k_transpose(const float* __restrict__ src,
                                                   unsigned short* __restrict__ dst,
                                                   int R, int C) {
  __shared__ float tile[32][33];
  const int e = blockIdx.z;
  const float* s = src + (size_t)e * R * C;
  unsigned short* d = dst + (size_t)e * R * C;
  const int c0 = blockIdx.x * 32, r0 = blockIdx.y * 32;
  const int tx = threadIdx.x & 31, ty = threadIdx.x >> 5;
#pragma unroll
  for (int i = ty; i < 32; i += 8)
    tile[i][tx] = s[(size_t)(r0 + i) * C + c0 + tx];
  __syncthreads();
#pragma unroll
  for (int i = ty; i < 32; i += 8)
    d[(size_t)(c0 + i) * R + r0 + tx] = f2bf(tile[tx][i]);
}

// ---------------- gating: one wave32 per token ----------------
__global__ void k_gate(const float* __restrict__ x, const float* __restrict__ gw,
                       const float* __restrict__ gb, int* __restrict__ topi,
                       float* __restrict__ topw, int* __restrict__ cnt) {
  int lane = threadIdx.x & 31;
  int tok = (blockIdx.x * blockDim.x + threadIdx.x) >> 5;
  if (tok >= NT) return;
  const float* xr = x + (size_t)tok * DM;
  float acc[NE];
#pragma unroll
  for (int e = 0; e < NE; e++) acc[e] = 0.f;
  for (int d = lane; d < DM; d += 32) {
    float xv = xr[d];
    const float4* g = (const float4*)(gw + (size_t)d * NE);
    float4 g0 = g[0], g1 = g[1];
    acc[0] += xv * g0.x; acc[1] += xv * g0.y; acc[2] += xv * g0.z; acc[3] += xv * g0.w;
    acc[4] += xv * g1.x; acc[5] += xv * g1.y; acc[6] += xv * g1.z; acc[7] += xv * g1.w;
  }
#pragma unroll
  for (int e = 0; e < NE; e++)
    for (int o = 16; o > 0; o >>= 1) acc[e] += __shfl_xor(acc[e], o, 32);
  if (lane == 0) {
    float s[NE], p[NE], mx = -1e30f, sum = 0.f;
#pragma unroll
    for (int e = 0; e < NE; e++) { s[e] = acc[e] + gb[e]; mx = fmaxf(mx, s[e]); }
#pragma unroll
    for (int e = 0; e < NE; e++) { p[e] = __expf(s[e] - mx); sum += p[e]; }
    int i0 = 0; float v0 = p[0];
#pragma unroll
    for (int e = 1; e < NE; e++) if (p[e] > v0) { v0 = p[e]; i0 = e; }
    int i1 = -1; float v1 = -1e30f;
#pragma unroll
    for (int e = 0; e < NE; e++) if (e != i0 && p[e] > v1) { v1 = p[e]; i1 = e; }
    float inv = 1.f / sum;
    topi[tok * 2] = i0; topi[tok * 2 + 1] = i1;
    topw[tok * 2] = v0 * inv; topw[tok * 2 + 1] = v1 * inv;
    atomicAdd(&cnt[i0], 1); atomicAdd(&cnt[i1], 1);
  }
}

// ---------------- prefix sum + load-balancing loss ----------------
__global__ void k_prefix(const int* __restrict__ cnt, int* __restrict__ off,
                         float* __restrict__ loss_out) {
  if (threadIdx.x == 0 && blockIdx.x == 0) {
    int o = 0;
    for (int e = 0; e < NE; e++) { off[e] = o; o += cnt[e]; }
    float tot = (float)o, l = 0.f;
    for (int e = 0; e < NE; e++) { float f = (float)cnt[e] / tot; l += f * f; }
    *loss_out = l;
  }
}

// ---------------- scatter pairs to expert-contiguous slots ----------------
__global__ void k_scatter(const int* __restrict__ topi, const int* __restrict__ off,
                          int* __restrict__ fill, int* __restrict__ slot_tok,
                          int* __restrict__ tok2slot) {
  int i = blockIdx.x * blockDim.x + threadIdx.x;
  if (i >= NT * TK) return;
  int e = topi[i];
  int pos = atomicAdd(&fill[e], 1);
  int s = off[e] + pos;
  slot_tok[s] = i >> 1;
  tok2slot[i] = s;
}

// ---------------- grouped GEMM1: h = relu(X_e @ w1[e] + b1[e]) ----------------
// B operand comes pre-transposed: w1t = [NE][DH][DM] bf16.
__global__ __launch_bounds__(256) void k_gemm1(
    const unsigned short* __restrict__ xb,   // [NT][DM] bf16
    const unsigned short* __restrict__ w1t,  // [NE][DH][DM] bf16 (transposed)
    const float* __restrict__ b1,            // [NE][DH]
    const int* __restrict__ cnt, const int* __restrict__ off,
    const int* __restrict__ slot_tok,
    unsigned short* __restrict__ h)          // [NT*TK][DH] bf16
{
  __shared__ __align__(16) unsigned short As[2][128 * 40];
  __shared__ __align__(16) unsigned short Bs[2][128 * 40];

  const int e = blockIdx.x >> 6;
  const int m0 = (blockIdx.x & 63) * 128;
  const int rows = cnt[e];
  if (m0 >= rows) return;
  const int n0 = blockIdx.y * 128;
  const int base = off[e];
  const unsigned short* wb = w1t + (size_t)e * DM * DH;

  const int tid = threadIdx.x;
  const int lane = tid & 31;
  const int wid = tid >> 5;
  const int m0w = (wid & 3) * 32;
  const int n0w = (wid >> 2) * 64;

  v8f zf = {0.f, 0.f, 0.f, 0.f, 0.f, 0.f, 0.f, 0.f};
  v8f acc[2][4];
#pragma unroll
  for (int i = 0; i < 2; i++)
#pragma unroll
    for (int j = 0; j < 4; j++) acc[i][j] = zf;

  // loader: thread -> (row = tid>>1, 32B chunk pair = tid&1)
  const int lr = tid >> 1, lc = tid & 1;
  int arow = m0 + lr; if (arow >= rows) arow = rows - 1;   // clamp (masked at store)
  const int atok = slot_tok[base + arow];
  const unsigned aoff = (unsigned)atok * (DM * 2) + lc * 32;        // bytes
  const unsigned boff = (unsigned)(n0 + lr) * (DM * 2) + lc * 32;   // row of w1t
  const unsigned ldsOffAB = lr * 80 + lc * 32;                      // bytes (stride 40 halves)
  const unsigned ldsA[2] = { ldsaddr(As[0]) + ldsOffAB, ldsaddr(As[1]) + ldsOffAB };
  const unsigned ldsB[2] = { ldsaddr(Bs[0]) + ldsOffAB, ldsaddr(Bs[1]) + ldsOffAB };

  const int l15 = lane & 15, hi = lane >> 4;
  const int NSTEP = DM / 32;

  // prologue: stage step 0 into buffer 0
  async_cp32(ldsA[0], xb, aoff);
  async_cp32(ldsB[0], wb, boff);

  for (int it = 0; it < NSTEP; ++it) {
    const int b = it & 1;
    if (it + 1 < NSTEP) {   // stage next step into other buffer
      const unsigned k2 = (unsigned)(it + 1) * 64;
      async_cp32(ldsA[1 - b], xb, aoff + k2);
      async_cp32(ldsB[1 - b], wb, boff + k2);
      asm volatile("s_wait_asynccnt 0x4" ::: "memory");
    } else {
      asm volatile("s_wait_asynccnt 0x0" ::: "memory");
    }
    __syncthreads();

    v16bf a[2], bfrag[4];
#pragma unroll
    for (int mf = 0; mf < 2; mf++) {   // A frag: K slots {8hi..}, {16+8hi..}
      int row = m0w + mf * 16 + l15;
      v8bf lo = *(const v8bf*)&As[b][row * 40 + hi * 8];
      v8bf hh = *(const v8bf*)&As[b][row * 40 + 16 + hi * 8];
      a[mf] = __builtin_shufflevector(lo, hh, 0, 1, 2, 3, 4, 5, 6, 7, 8, 9, 10, 11, 12, 13, 14, 15);
    }
#pragma unroll
    for (int nf = 0; nf < 4; nf++) {   // B frag: K slots {8hi..}, {16+8hi..}
      int col = n0w + nf * 16 + l15;
      v8bf lo = *(const v8bf*)&Bs[b][col * 40 + hi * 8];
      v8bf hh = *(const v8bf*)&Bs[b][col * 40 + 16 + hi * 8];
      bfrag[nf] = __builtin_shufflevector(lo, hh, 0, 1, 2, 3, 4, 5, 6, 7, 8, 9, 10, 11, 12, 13, 14, 15);
    }
#pragma unroll
    for (int mf = 0; mf < 2; mf++)
#pragma unroll
      for (int nf = 0; nf < 4; nf++)
        acc[mf][nf] = __builtin_amdgcn_wmma_f32_16x16x32_bf16(
            false, a[mf], false, bfrag[nf], (short)0, acc[mf][nf], false, false);
    __syncthreads();
  }

#pragma unroll
  for (int mf = 0; mf < 2; mf++)
#pragma unroll
    for (int nf = 0; nf < 4; nf++) {
      int col = n0 + n0w + nf * 16 + l15;
      float bias = b1[(size_t)e * DH + col];
      int rbase = m0 + m0w + mf * 16 + hi * 8;
#pragma unroll
      for (int r = 0; r < 8; r++) {
        int row = rbase + r;
        if (row < rows) {
          float v = acc[mf][nf][r] + bias;
          v = v > 0.f ? v : 0.f;
          h[(size_t)(base + row) * DH + col] = f2bf(v);
        }
      }
    }
}

// ---------------- grouped GEMM2: y = H_e @ w2[e] + b2[e] ----------------
// B operand pre-transposed: w2t = [NE][DM][DH] bf16.
__global__ __launch_bounds__(256) void k_gemm2(
    const unsigned short* __restrict__ h,    // [NT*TK][DH] bf16
    const unsigned short* __restrict__ w2t,  // [NE][DM][DH] bf16 (transposed)
    const float* __restrict__ b2,            // [NE][DM]
    const int* __restrict__ cnt, const int* __restrict__ off,
    float* __restrict__ y)                   // [NT*TK][DM] fp32
{
  __shared__ __align__(16) unsigned short As[2][128 * 40];
  __shared__ __align__(16) unsigned short Bs[2][128 * 40];

  const int e = blockIdx.x >> 6;
  const int m0 = (blockIdx.x & 63) * 128;
  const int rows = cnt[e];
  if (m0 >= rows) return;
  const int n0 = blockIdx.y * 128;
  const int base = off[e];
  const unsigned short* wb = w2t + (size_t)e * DM * DH;

  const int tid = threadIdx.x;
  const int lane = tid & 31;
  const int wid = tid >> 5;
  const int m0w = (wid & 3) * 32;
  const int n0w = (wid >> 2) * 64;

  v8f zf = {0.f, 0.f, 0.f, 0.f, 0.f, 0.f, 0.f, 0.f};
  v8f acc[2][4];
#pragma unroll
  for (int i = 0; i < 2; i++)
#pragma unroll
    for (int j = 0; j < 4; j++) acc[i][j] = zf;

  const int lr = tid >> 1, lc = tid & 1;
  int arow = m0 + lr; if (arow >= rows) arow = rows - 1;
  const unsigned aoff = (unsigned)(base + arow) * (DH * 2) + lc * 32;
  const unsigned boff = (unsigned)(n0 + lr) * (DH * 2) + lc * 32;   // row of w2t
  const unsigned ldsOffAB = lr * 80 + lc * 32;
  const unsigned ldsA[2] = { ldsaddr(As[0]) + ldsOffAB, ldsaddr(As[1]) + ldsOffAB };
  const unsigned ldsB[2] = { ldsaddr(Bs[0]) + ldsOffAB, ldsaddr(Bs[1]) + ldsOffAB };

  const int l15 = lane & 15, hi = lane >> 4;
  const int NSTEP = DH / 32;

  async_cp32(ldsA[0], h, aoff);
  async_cp32(ldsB[0], wb, boff);

  for (int it = 0; it < NSTEP; ++it) {
    const int b = it & 1;
    if (it + 1 < NSTEP) {
      const unsigned k2 = (unsigned)(it + 1) * 64;
      async_cp32(ldsA[1 - b], h, aoff + k2);
      async_cp32(ldsB[1 - b], wb, boff + k2);
      asm volatile("s_wait_asynccnt 0x4" ::: "memory");
    } else {
      asm volatile("s_wait_asynccnt 0x0" ::: "memory");
    }
    __syncthreads();

    v16bf a[2], bfrag[4];
#pragma unroll
    for (int mf = 0; mf < 2; mf++) {
      int row = m0w + mf * 16 + l15;
      v8bf lo = *(const v8bf*)&As[b][row * 40 + hi * 8];
      v8bf hh = *(const v8bf*)&As[b][row * 40 + 16 + hi * 8];
      a[mf] = __builtin_shufflevector(lo, hh, 0, 1, 2, 3, 4, 5, 6, 7, 8, 9, 10, 11, 12, 13, 14, 15);
    }
#pragma unroll
    for (int nf = 0; nf < 4; nf++) {
      int col = n0w + nf * 16 + l15;
      v8bf lo = *(const v8bf*)&Bs[b][col * 40 + hi * 8];
      v8bf hh = *(const v8bf*)&Bs[b][col * 40 + 16 + hi * 8];
      bfrag[nf] = __builtin_shufflevector(lo, hh, 0, 1, 2, 3, 4, 5, 6, 7, 8, 9, 10, 11, 12, 13, 14, 15);
    }
#pragma unroll
    for (int mf = 0; mf < 2; mf++)
#pragma unroll
      for (int nf = 0; nf < 4; nf++)
        acc[mf][nf] = __builtin_amdgcn_wmma_f32_16x16x32_bf16(
            false, a[mf], false, bfrag[nf], (short)0, acc[mf][nf], false, false);
    __syncthreads();
  }

#pragma unroll
  for (int mf = 0; mf < 2; mf++)
#pragma unroll
    for (int nf = 0; nf < 4; nf++) {
      int col = n0 + n0w + nf * 16 + l15;
      float bias = b2[(size_t)e * DM + col];
      int rbase = m0 + m0w + mf * 16 + hi * 8;
#pragma unroll
      for (int r = 0; r < 8; r++) {
        int row = rbase + r;
        if (row < rows)
          y[(size_t)(base + row) * DM + col] = acc[mf][nf][r] + bias;
      }
    }
}

// ---------------- combine top-2 expert outputs ----------------
__global__ void k_combine(const float* __restrict__ y, const int* __restrict__ t2s,
                          const float* __restrict__ topw, float* __restrict__ out) {
  int tok = blockIdx.x;
  int s0 = t2s[tok * 2], s1 = t2s[tok * 2 + 1];
  float w0 = topw[tok * 2], w1 = topw[tok * 2 + 1];
  const float* y0 = y + (size_t)s0 * DM;
  const float* y1 = y + (size_t)s1 * DM;
  float* o = out + (size_t)tok * DM;
  for (int d = threadIdx.x; d < DM; d += blockDim.x)
    o[d] = w0 * y0[d] + w1 * y1[d];
}

extern "C" void kernel_launch(void* const* d_in, const int* in_sizes, int n_in,
                              void* d_out, int out_size, void* d_ws, size_t ws_size,
                              hipStream_t stream) {
  (void)in_sizes; (void)n_in; (void)out_size; (void)ws_size;
  const float* x  = (const float*)d_in[0];
  const float* gw = (const float*)d_in[1];
  const float* gb = (const float*)d_in[2];
  const float* w1 = (const float*)d_in[3];
  const float* b1 = (const float*)d_in[4];
  const float* w2 = (const float*)d_in[5];
  const float* b2 = (const float*)d_in[6];
  float* out = (float*)d_out;

  char* ws = (char*)d_ws;
  size_t o = 0;
  auto alloc = [&](size_t bytes) -> char* {
    char* p = ws + o;
    o += (bytes + 255) & ~(size_t)255;
    return p;
  };
  unsigned short* xb  = (unsigned short*)alloc((size_t)NT * DM * 2);
  unsigned short* w1t = (unsigned short*)alloc((size_t)NE * DM * DH * 2);
  unsigned short* w2t = (unsigned short*)alloc((size_t)NE * DM * DH * 2);
  unsigned short* h   = (unsigned short*)alloc((size_t)NT * TK * DH * 2);
  float* y            = (float*)alloc((size_t)NT * TK * DM * 4);
  int*   topi = (int*)alloc(NT * TK * 4);
  float* topw = (float*)alloc(NT * TK * 4);
  int*   t2s  = (int*)alloc(NT * TK * 4);
  int*   stok = (int*)alloc(NT * TK * 4);
  int*   cnt  = (int*)alloc(64);
  int*   offp = (int*)alloc(64);
  int*   fill = (int*)alloc(64);

  k_init<<<1, 32, 0, stream>>>(cnt, fill);
  k_convert<<<2048, 256, 0, stream>>>(x, xb, NT * DM);
  // w1 [E][DM][DH] -> w1t [E][DH][DM]; w2 [E][DH][DM] -> w2t [E][DM][DH]
  k_transpose<<<dim3(DH / 32, DM / 32, NE), 256, 0, stream>>>(w1, w1t, DM, DH);
  k_transpose<<<dim3(DM / 32, DH / 32, NE), 256, 0, stream>>>(w2, w2t, DH, DM);
  k_gate<<<NT / 8, 256, 0, stream>>>(x, gw, gb, topi, topw, cnt);
  k_prefix<<<1, 1, 0, stream>>>(cnt, offp, out + (size_t)NT * DM);
  k_scatter<<<(NT * TK) / 256, 256, 0, stream>>>(topi, offp, fill, stok, t2s);
  k_gemm1<<<dim3(512, 32), 256, 0, stream>>>(xb, w1t, b1, cnt, offp, stok, h);
  k_gemm2<<<dim3(512, 8), 256, 0, stream>>>(h, w2t, b2, cnt, offp, y);
  k_combine<<<NT, 256, 0, stream>>>(y, t2s, topw, out);
}
